// Attention_48653389529729
// MI455X (gfx1250) — matrix operand order repaired and checked by reading, not compiled, not run on hardware
//
#include <hip/hip_runtime.h>

typedef _Float16 v8h  __attribute__((ext_vector_type(8)));
typedef _Float16 v16h __attribute__((ext_vector_type(16)));
typedef float    v8f  __attribute__((ext_vector_type(8)));

#define HDIM 500        // H
#define HP   512        // padded H (N dim and K dim)
#define SDIM 512        // S
#define BDIM 256        // B
#define ASTR 520        // f16 LDS row stride (520/2 = 260 ≡ 4 mod 64 -> conflict-free b128)

// ---------------------------------------------------------------------------
// Kernel 1: W1t[k][512] f32 (transposed, for hidden-part GEMM) and
//           W2h[h][512] f16 (original orientation, zero-padded both dims)
// ---------------------------------------------------------------------------
__global__ void prep_weights(const float* __restrict__ W,
                             float* __restrict__ W1t, _Float16* __restrict__ W2h) {
    int idx = blockIdx.x * blockDim.x + threadIdx.x;   // 512*512
    int k = idx & 511;              // fastest -> coalesced along W rows
    int h = idx >> 9;
    bool hv = (h < HDIM), kv = (k < HDIM);
    if (kv) W1t[k * HP + h] = hv ? W[h * (2 * HDIM) + k] : 0.f;
    W2h[h * HP + k] = (hv && kv) ? (_Float16)W[h * (2 * HDIM) + HDIM + k] : (_Float16)0.f;
}

// ---------------------------------------------------------------------------
// Kernel 2: hidden_part[b][h] = b_attn[h] + hidden[b,:] . W1[h,:]  (f32, h-padded)
// ---------------------------------------------------------------------------
__global__ void prep_hidden(const float* __restrict__ hidden,
                            const float* __restrict__ b_attn,
                            const float* __restrict__ W1t,
                            float* __restrict__ hp) {
    __shared__ float hrow[HDIM];
    const int b = blockIdx.x;
    const int t = threadIdx.x;                 // 256
    for (int i = t; i < HDIM; i += 256) hrow[i] = hidden[b * HDIM + i];
    __syncthreads();
    float acc0 = 0.f, acc1 = 0.f;
    const int h0 = t, h1 = t + 256;
    for (int k = 0; k < HDIM; ++k) {
        float hv = hrow[k];
        acc0 = fmaf(hv, W1t[k * HP + h0], acc0);
        acc1 = fmaf(hv, W1t[k * HP + h1], acc1);
    }
    hp[b * HP + h0] = acc0 + b_attn[h0];
    hp[b * HP + h1] = (h1 < HDIM) ? (acc1 + b_attn[h1]) : 0.f;
}

static __device__ __forceinline__ unsigned pack2h(float a, float b) {
    _Float16 ha = (_Float16)a, hb = (_Float16)b;
    return (unsigned)__builtin_bit_cast(unsigned short, ha) |
           ((unsigned)__builtin_bit_cast(unsigned short, hb) << 16);
}

static __device__ __forceinline__ v16h ld32h(const _Float16* p) {
    v8h lo = *(const v8h*)(p);          // 16B aligned ds_load_b128
    v8h hi = *(const v8h*)(p + 8);
    return __builtin_shufflevector(lo, hi, 0,1,2,3,4,5,6,7,8,9,10,11,12,13,14,15);
}

// ---------------------------------------------------------------------------
// Kernel 3: scores[s,b] = v . tanh(hp[b,:] + enc[s,b,:] @ W2^T), f16 WMMA.
// 128 threads = 4 waves; wave owns a 16-row M-tile; N processed in groups of 4
// tiles so each A fragment feeds 4 independent WMMAs (2.5 b128 loads / WMMA).
// ---------------------------------------------------------------------------
__global__ void attn_scores(const float* __restrict__ enc,
                            const _Float16* __restrict__ W2h,
                            const float* __restrict__ hp,
                            const float* __restrict__ v,
                            float* __restrict__ scores) {
    __shared__ _Float16 As[64 * ASTR];   // 64 rows x 520 f16  (66,560 B)
    __shared__ _Float16 Bs[64 * ASTR];   // 4 N-tiles (64 n-rows) x 520 f16 (66,560 B)

    const int t      = threadIdx.x;
    const int wave   = t >> 5;
    const int lane   = t & 31;
    const int laneHi = lane >> 4;
    const int nl     = lane & 15;

    const long m0    = (long)blockIdx.x * 64;     // flat row = s*B + b
    const int  bBase = (int)(m0 & (BDIM - 1));
    const int  sIdx  = (int)(m0 >> 8);

    // --- stage A: 64 contiguous rows (64*500 f32, contiguous in HBM), cvt->f16 ---
    {
        const float4* src = (const float4*)(enc + (size_t)m0 * HDIM);
        for (int i = t; i < 64 * 125; i += 128) {
            int row = i / 125, c4 = i % 125;
            float4 val = src[i];
            uint2 pk;
            pk.x = pack2h(val.x, val.y);
            pk.y = pack2h(val.z, val.w);
            *(uint2*)(&As[row * ASTR + c4 * 4]) = pk;
        }
        for (int i = t; i < 64 * 12; i += 128) {       // zero K-pad cols 500..511
            int row = i / 12, c = i % 12;
            As[row * ASTR + HDIM + c] = (_Float16)0.f;
        }
    }

    float acc[8] = {0.f, 0.f, 0.f, 0.f, 0.f, 0.f, 0.f, 0.f};
    const _Float16* Arow = &As[(wave * 16 + nl) * ASTR];
    const int mrow = (bBase + wave * 16 + laneHi * 8) * HP;   // hp row base (+r*HP)

    for (int g = 0; g < 8; ++g) {                 // 8 groups x 4 N-tiles = 512 cols
        const int n0 = g * 64;
        __syncthreads();                          // Bs reuse fence (fences As on g==0)
        // stage B group: W2h rows n0..n0+63, full padded K (f16, no conversion)
        for (int i = t; i < 64 * 64; i += 128) {
            int ln = i >> 6, c8 = i & 63;
            uint4 w = *(const uint4*)(W2h + (size_t)(n0 + ln) * HP + c8 * 8);
            *(uint4*)(&Bs[ln * ASTR + c8 * 8]) = w;
        }
        __syncthreads();

        // C init with s-invariant hidden part (full f32)
        v8f c0, c1, c2, c3;
#pragma unroll
        for (int r = 0; r < 8; ++r) {
            c0[r] = hp[mrow + r * HP + n0      + nl];
            c1[r] = hp[mrow + r * HP + n0 + 16 + nl];
            c2[r] = hp[mrow + r * HP + n0 + 32 + nl];
            c3[r] = hp[mrow + r * HP + n0 + 48 + nl];
        }

        for (int k0 = 0; k0 < HP; k0 += 32) {
            // A fragment: elems 0..15 = K k0+16*laneHi .. +15 (contiguous, matches B)
            v8h alo = *(const v8h*)(Arow + k0 + 16 * laneHi);
            v8h ahi = *(const v8h*)(Arow + k0 + 16 * laneHi + 8);
            v16h a = __builtin_shufflevector(alo, ahi,
                        0,1,2,3,4,5,6,7,8,9,10,11,12,13,14,15);
            // B fragments: 16 consecutive K at row n=nl of each tile, chunk by laneHi
            const int bcol = k0 + 16 * laneHi;
            v16h b0 = ld32h(&Bs[(     nl) * ASTR + bcol]);
            v16h b1 = ld32h(&Bs[(16 + nl) * ASTR + bcol]);
            v16h b2 = ld32h(&Bs[(32 + nl) * ASTR + bcol]);
            v16h b3 = ld32h(&Bs[(48 + nl) * ASTR + bcol]);
            c0 = __builtin_amdgcn_wmma_f32_16x16x32_f16(false, a, false, b0, (short)0, c0, false, false);
            c1 = __builtin_amdgcn_wmma_f32_16x16x32_f16(false, a, false, b1, (short)0, c1, false, false);
            c2 = __builtin_amdgcn_wmma_f32_16x16x32_f16(false, a, false, b2, (short)0, c2, false, false);
            c3 = __builtin_amdgcn_wmma_f32_16x16x32_f16(false, a, false, b3, (short)0, c3, false, false);
        }

        // fused epilogue: tanh then dot with v (v==0 masks padded h>=500)
        const int h0c = n0 + nl;
        float vh0 = (h0c      < HDIM) ? v[h0c]      : 0.f;
        float vh1 = (h0c + 16 < HDIM) ? v[h0c + 16] : 0.f;
        float vh2 = (h0c + 32 < HDIM) ? v[h0c + 32] : 0.f;
        float vh3 = (h0c + 48 < HDIM) ? v[h0c + 48] : 0.f;
#pragma unroll
        for (int r = 0; r < 8; ++r) {
            acc[r] += tanhf(c0[r]) * vh0 + tanhf(c1[r]) * vh1
                    + tanhf(c2[r]) * vh2 + tanhf(c3[r]) * vh3;
        }
    }

    // reduce each row's partial dot across its 16-lane N group
#pragma unroll
    for (int r = 0; r < 8; ++r) {
        float x = acc[r];
        x += __shfl_xor(x, 1, 32);
        x += __shfl_xor(x, 2, 32);
        x += __shfl_xor(x, 4, 32);
        x += __shfl_xor(x, 8, 32);
        acc[r] = x;
    }
    if (nl == 0) {                   // lanes 0 (rows r) and 16 (rows r+8)
#pragma unroll
        for (int r = 0; r < 8; ++r) {
            int b = bBase + wave * 16 + r + laneHi * 8;
            scores[b * SDIM + sIdx] = acc[r];
        }
    }
}

// ---------------------------------------------------------------------------
// Kernel 4: softmax over S for each b; out[b,0,s]
// ---------------------------------------------------------------------------
__global__ void softmax_rows(const float* __restrict__ sc, float* __restrict__ out) {
    __shared__ float redm[8];
    __shared__ float reds[8];
    const int b = blockIdx.x;
    const int t = threadIdx.x;
    float x0 = sc[b * SDIM + t];
    float x1 = sc[b * SDIM + t + 256];
    float m = fmaxf(x0, x1);
#pragma unroll
    for (int d = 16; d >= 1; d >>= 1) m = fmaxf(m, __shfl_xor(m, d, 32));
    if ((t & 31) == 0) redm[t >> 5] = m;
    __syncthreads();
    m = redm[0];
#pragma unroll
    for (int i = 1; i < 8; ++i) m = fmaxf(m, redm[i]);
    float e0 = expf(x0 - m), e1 = expf(x1 - m);
    float s = e0 + e1;
#pragma unroll
    for (int d = 16; d >= 1; d >>= 1) s += __shfl_xor(s, d, 32);
    if ((t & 31) == 0) reds[t >> 5] = s;
    __syncthreads();
    float tot = reds[0];
#pragma unroll
    for (int i = 1; i < 8; ++i) tot += reds[i];
    float inv = 1.0f / tot;
    out[b * SDIM + t]       = e0 * inv;
    out[b * SDIM + t + 256] = e1 * inv;
}

// ---------------------------------------------------------------------------
extern "C" void kernel_launch(void* const* d_in, const int* in_sizes, int n_in,
                              void* d_out, int out_size, void* d_ws, size_t ws_size,
                              hipStream_t stream) {
    const float* hidden = (const float*)d_in[0];   // [1,B,H]
    const float* enc    = (const float*)d_in[1];   // [S,B,H]
    const float* W      = (const float*)d_in[2];   // [H,2H]
    const float* b_attn = (const float*)d_in[3];   // [H]
    const float* v      = (const float*)d_in[4];   // [H]
    float* out = (float*)d_out;                    // [B,1,S]
    float* ws  = (float*)d_ws;

    float*     W1t = ws;                           // 500*512 f32   = 256000 floats
    float*     hp  = ws + 256000;                  // 256*512 f32   = 131072 floats
    float*     sc  = ws + 387072;                  // 256*512 f32   = 131072 floats
    _Float16*  W2h = (_Float16*)(ws + 518144);     // 512*512 f16   = 131072 float-slots
                                                   // total ~2.6 MB of workspace

    prep_weights<<<1024, 256, 0, stream>>>(W, W1t, W2h);
    prep_hidden <<<BDIM, 256, 0, stream>>>(hidden, b_attn, W1t, hp);
    attn_scores <<<2048, 128, 0, stream>>>(enc, W2h, hp, v, sc);
    softmax_rows<<<BDIM, 256, 0, stream>>>(sc, out);
}
